// MultiHeadSelfAttention_5892695130267
// MI455X (gfx1250) — matrix operand order repaired and checked by reading, not compiled, not run on hardware
//
#include <hip/hip_runtime.h>
#include <hip/hip_bf16.h>

// ---------------------------------------------------------------------------
// MHSA for MI455X (gfx1250): bf16 WMMA GEMMs + cooperative flash attention.
// B=2, S=2048, D=1024, H=16, HD=64.  f32 accumulation throughout.
// ---------------------------------------------------------------------------

typedef __attribute__((ext_vector_type(16))) __bf16 v16bf;
typedef __attribute__((ext_vector_type(8)))  __bf16 v8bf;
typedef __attribute__((ext_vector_type(8)))  float  v8f;
typedef __attribute__((ext_vector_type(4)))  int    v4i;

static constexpr int Bb = 2, Ss = 2048, Dd = 1024, Hh = 16, HD = 64;
static constexpr int Mrows = Bb * Ss; // 4096

// float -> bf16 bits (round to nearest even)
__device__ __forceinline__ unsigned short f2bf(float f) {
  unsigned u = __float_as_uint(f);
  u += 0x7fffu + ((u >> 16) & 1u);
  return (unsigned short)(u >> 16);
}

__device__ __forceinline__ v8bf ld8(const unsigned short* p) {
  return *reinterpret_cast<const v8bf*>(p);
}

__device__ __forceinline__ v16bf combine16(v8bf lo, v8bf hi) {
  v16bf r;
#pragma unroll
  for (int i = 0; i < 8; ++i) { r[i] = lo[i]; r[i + 8] = hi[i]; }
  return r;
}

// A fragment (16x32 bf16, MxK): lane row = lane&15; elems 0..7 = K kb..kb+7,
// elems 8..15 = K kb+16..kb+23, kb = k0 + (lane>=16 ? 8 : 0).
__device__ __forceinline__ v16bf ld_afrag(const unsigned short* rowptr, int kb) {
  return combine16(ld8(rowptr + kb), ld8(rowptr + kb + 16));
}

// B fragment (32x16 bf16, KxN) from row-major B^T: lane col n = lane&15,
// elems 0..7 = Bt[n][kb2..kb2+7], elems 8..15 = Bt[n][kb2+16..kb2+23],
// kb2 = k0 + (lane>=16 ? 8 : 0). p = &Bt[n][kb2].
__device__ __forceinline__ v16bf ld_bfrag(const unsigned short* p) {
  return combine16(ld8(p), ld8(p + 16));
}

__device__ __forceinline__ v8f wmma_bf16(v16bf a, v16bf b, v8f c) {
  return __builtin_amdgcn_wmma_f32_16x16x32_bf16(false, a, false, b,
                                                 (short)0, c, false, false);
}

// ---- CDNA5 async global->LDS staging (16 bytes per lane) -------------------
#if __has_builtin(__builtin_amdgcn_global_load_async_to_lds_b128)
#define HAS_ASYNC_LDS 1
#else
#define HAS_ASYNC_LDS 0
#endif

__device__ __forceinline__ void stage16(const unsigned short* g, unsigned short* l) {
#if HAS_ASYNC_LDS
  __builtin_amdgcn_global_load_async_to_lds_b128(
      (__attribute__((address_space(1))) v4i*)g,
      (__attribute__((address_space(3))) v4i*)l, 0, 0);
#else
  *reinterpret_cast<v8bf*>(l) = *reinterpret_cast<const v8bf*>(g);
#endif
}

__device__ __forceinline__ void async_wait_all() {
#if __has_builtin(__builtin_amdgcn_s_wait_asynccnt)
  __builtin_amdgcn_s_wait_asynccnt(0);
#elif HAS_ASYNC_LDS
  asm volatile("s_wait_asynccnt 0" ::: "memory");
#endif
}

// ---------------------------------------------------------------------------
// fp32 -> bf16 elementwise
// ---------------------------------------------------------------------------
__global__ __launch_bounds__(256) void cvt_bf16_kernel(const float* __restrict__ src,
                                                       unsigned short* __restrict__ dst,
                                                       int n) {
  int i = blockIdx.x * 256 + threadIdx.x;
  if (i < n) dst[i] = f2bf(src[i]);
}

// W [K=1024][N=1024] fp32 -> Wt [N][K] bf16
__global__ __launch_bounds__(256) void cvt_wt_kernel(const float* __restrict__ W,
                                                     unsigned short* __restrict__ Wt) {
  int i = blockIdx.x * 256 + threadIdx.x;   // i < 1024*1024
  int k = i >> 10, n = i & 1023;
  Wt[n * 1024 + k] = f2bf(W[i]);
}

// ---------------------------------------------------------------------------
// GEMM: Y[4096][1024] = A (bf16) @ Bt^T (bf16) + bias; wave -> 16x64 strip.
// MODE 0: bf16 out [B,H,S,HD] (Q,K) | MODE 1: bf16 out [B,H,HD,S] (V^T)
// MODE 2: f32 out row-major (final O projection)
// ---------------------------------------------------------------------------
template <int MODE>
__global__ __launch_bounds__(256) void gemm_kernel(const unsigned short* __restrict__ A,
                                                   const unsigned short* __restrict__ Bt,
                                                   const float* __restrict__ bias,
                                                   void* __restrict__ dst) {
  const int lane = threadIdx.x & 31;
  const int wave = threadIdx.x >> 5;
  const int gw = blockIdx.x * 8 + wave;
  const int nstrip = gw & 15;
  const int mtile  = gw >> 4;
  const int lo15 = lane & 15;
  const int hi8  = (lane & 16) ? 8 : 0;

  const unsigned short* arow = A + (size_t)(mtile * 16 + lo15) * 1024;
  const int akb = (lane & 16) ? 8 : 0;
  const int bkb = (lane & 16) ? 8 : 0;

  v8f c[4];
#pragma unroll
  for (int g = 0; g < 4; ++g)
#pragma unroll
    for (int j = 0; j < 8; ++j) c[g][j] = 0.0f;

  for (int k0 = 0; k0 < 1024; k0 += 32) {
    v16bf a = ld_afrag(arow, k0 + akb);
#pragma unroll
    for (int g = 0; g < 4; ++g) {
      int n = nstrip * 64 + g * 16 + lo15;
      v16bf b = ld_bfrag(Bt + (size_t)n * 1024 + k0 + bkb);
      c[g] = wmma_bf16(a, b, c[g]);
    }
  }

#pragma unroll
  for (int g = 0; g < 4; ++g) {
    int n = nstrip * 64 + g * 16 + lo15;
    float bv = bias[n];
#pragma unroll
    for (int j = 0; j < 8; ++j) {
      int m = mtile * 16 + j + hi8;
      float val = c[g][j] + bv;
      if (MODE == 2) {
        ((float*)dst)[(size_t)m * 1024 + n] = val;
      } else {
        int b = m >> 11, s = m & 2047;
        int h = n >> 6,  hd = n & 63;
        size_t idx = (MODE == 0)
          ? (((size_t)(b * Hh + h) * Ss + s) * HD + hd)
          : (((size_t)(b * Hh + h) * HD + hd) * Ss + s);
        ((unsigned short*)dst)[idx] = f2bf(val);
      }
    }
  }
}

// ---------------------------------------------------------------------------
// Cooperative flash attention.
// Block = 8 waves = 8 query tiles (128 queries) of one (b,h). Keys processed
// 64 at a time; K/V slabs staged into LDS by the whole block (async global->
// LDS when available, double-buffered so DMA overlaps WMMA compute).
// Base-2 online softmax: logits pre-scaled by (1/sqrt(64))*log2(e).
// Q,K: [B,H,S,64] bf16.  Vt: [B,H,64,S] bf16.  vals out: [B,S,D] bf16.
// ---------------------------------------------------------------------------
__global__ __launch_bounds__(256) void attn_kernel(const unsigned short* __restrict__ Q,
                                                   const unsigned short* __restrict__ K,
                                                   const unsigned short* __restrict__ Vt,
                                                   unsigned short* __restrict__ vals) {
  __shared__ __align__(16) unsigned short ksm[2][64 * 64];  // [key][hd]   16 KB
  __shared__ __align__(16) unsigned short vsm[2][64 * 64];  // [hd][key]   16 KB
  __shared__ __align__(16) unsigned short psm[8][16 * 64];  // P per wave  16 KB

  const int tid  = threadIdx.x;
  const int lane = tid & 31;
  const int wave = tid >> 5;
  const int bh   = blockIdx.x >> 4;        // 32 heads
  const int qblk = blockIdx.x & 15;        // 16 query blocks of 128
  const int qt   = qblk * 8 + wave;        // this wave's 16-query tile
  const int bidx = bh >> 4, h = bh & 15;

  const int lo15 = lane & 15;
  const int hi8  = (lane & 16) ? 8 : 0;
  const int akb  = (lane & 16) ? 8 : 0;
  const int bkb  = (lane & 16) ? 8 : 0;

  const unsigned short* q  = Q  + (size_t)bh * Ss * HD;
  const unsigned short* k  = K  + (size_t)bh * Ss * HD;
  const unsigned short* vt = Vt + (size_t)bh * HD * Ss;

  // Q A-fragments (two 32-wide K-halves of HD=64), loaded once
  const unsigned short* qrow = q + (size_t)(qt * 16 + lo15) * HD;
  v16bf qa0 = ld_afrag(qrow, 0 + akb);
  v16bf qa1 = ld_afrag(qrow, 32 + akb);

  const float kScale = 0.125f * 1.44269504088896340736f;  // (1/sqrt(64))*log2e

  float m_r[8], l_r[8];
  v8f acc[4];
#pragma unroll
  for (int j = 0; j < 8; ++j) { m_r[j] = -1e30f; l_r[j] = 0.0f; }
#pragma unroll
  for (int g = 0; g < 4; ++g)
#pragma unroll
    for (int j = 0; j < 8; ++j) acc[g][j] = 0.0f;

  // --- block-cooperative staging of one 64-key slab (8 KB K + 8 KB Vt) ----
  auto stage_slab = [&](int kt, int buf) {
#pragma unroll
    for (int i = 0; i < 2; ++i) {                 // K: 512 x 16B chunks
      int c = tid + i * 256;
      stage16(k + (size_t)kt * 64 + c * 8, &ksm[buf][c * 8]);
    }
#pragma unroll
    for (int i = 0; i < 2; ++i) {                 // Vt: rows of 64 keys
      int c = tid + i * 256;
      int hd = c >> 3, cc = c & 7;
      stage16(vt + (size_t)hd * Ss + kt + cc * 8, &vsm[buf][hd * 64 + cc * 8]);
    }
  };

  stage_slab(0, 0);

  for (int kt = 0; kt < Ss; kt += 64) {
    const int cur = (kt >> 6) & 1;
    async_wait_all();
    __syncthreads();                       // slab[cur] ready; prev slab free
    if (kt + 64 < Ss) stage_slab(kt + 64, cur ^ 1);   // overlap DMA w/ compute

    // ---- logits: four 16x16 f32 fragments over this 64-key slab ----
    v8f sfr[4];
#pragma unroll
    for (int t = 0; t < 4; ++t) {
      v8f cz;
#pragma unroll
      for (int j = 0; j < 8; ++j) cz[j] = 0.0f;
      const unsigned short* krow = &ksm[cur][(t * 16 + lo15) * 64];
      cz = wmma_bf16(qa0, ld_bfrag(krow + 0 + bkb), cz);
      cz = wmma_bf16(qa1, ld_bfrag(krow + 32 + bkb), cz);
#pragma unroll
      for (int j = 0; j < 8; ++j) cz[j] *= kScale;
      sfr[t] = cz;
    }

    // ---- online softmax (base 2): row max / alpha / exp2 / row sum ----
    float rmax[8];
#pragma unroll
    for (int j = 0; j < 8; ++j)
      rmax[j] = fmaxf(fmaxf(sfr[0][j], sfr[1][j]), fmaxf(sfr[2][j], sfr[3][j]));
#pragma unroll
    for (int off = 1; off <= 8; off <<= 1)
#pragma unroll
      for (int j = 0; j < 8; ++j)
        rmax[j] = fmaxf(rmax[j], __shfl_xor(rmax[j], off, 32));

    float alpha[8];
#pragma unroll
    for (int j = 0; j < 8; ++j) {
      float mn = fmaxf(m_r[j], rmax[j]);
      alpha[j] = exp2f(m_r[j] - mn);
      m_r[j] = mn;
    }

    float rsum[8];
#pragma unroll
    for (int j = 0; j < 8; ++j) rsum[j] = 0.0f;
#pragma unroll
    for (int t = 0; t < 4; ++t)
#pragma unroll
      for (int j = 0; j < 8; ++j) {
        float p = exp2f(sfr[t][j] - m_r[j]);
        rsum[j] += p;
        psm[wave][(j + hi8) * 64 + t * 16 + lo15] = f2bf(p);  // C-layout -> LDS
      }
#pragma unroll
    for (int off = 1; off <= 8; off <<= 1)
#pragma unroll
      for (int j = 0; j < 8; ++j) rsum[j] += __shfl_xor(rsum[j], off, 32);
#pragma unroll
    for (int j = 0; j < 8; ++j) l_r[j] = l_r[j] * alpha[j] + rsum[j];

    // pre-scale accumulators by alpha (row mapping matches C layout)
#pragma unroll
    for (int g = 0; g < 4; ++g)
#pragma unroll
      for (int j = 0; j < 8; ++j) acc[g][j] *= alpha[j];

    __builtin_amdgcn_wave_barrier();       // order psm store -> load (same wave)

    // ---- P x V: two A-frags (keys 0..31 / 32..63), 4 hd-tiles each ----
    const unsigned short* prow = &psm[wave][lo15 * 64];
    v16bf pa0 = combine16(ld8(prow + akb),      ld8(prow + akb + 16));
    v16bf pa1 = combine16(ld8(prow + 32 + akb), ld8(prow + 32 + akb + 16));
#pragma unroll
    for (int g = 0; g < 4; ++g) {
      const unsigned short* vrow = &vsm[cur][(g * 16 + lo15) * 64];
      acc[g] = wmma_bf16(pa0, ld_bfrag(vrow + 0 + bkb), acc[g]);
      acc[g] = wmma_bf16(pa1, ld_bfrag(vrow + 32 + bkb), acc[g]);
    }
  }

  // ---- normalize and write to [B,S,D] bf16 ----
#pragma unroll
  for (int g = 0; g < 4; ++g) {
#pragma unroll
    for (int j = 0; j < 8; ++j) {
      int s = qt * 16 + j + hi8;
      int d = h * 64 + g * 16 + lo15;
      vals[((size_t)(bidx * Ss + s)) * Dd + d] = f2bf(acc[g][j] / l_r[j]);
    }
  }
}

// ---------------------------------------------------------------------------
extern "C" void kernel_launch(void* const* d_in, const int* in_sizes, int n_in,
                              void* d_out, int out_size, void* d_ws, size_t ws_size,
                              hipStream_t stream) {
  const float* x  = (const float*)d_in[0];
  const float* Wq = (const float*)d_in[1];
  const float* bq = (const float*)d_in[2];
  const float* Wk = (const float*)d_in[3];
  const float* bk = (const float*)d_in[4];
  const float* Wv = (const float*)d_in[5];
  const float* bv = (const float*)d_in[6];
  const float* Wo = (const float*)d_in[7];
  const float* bo = (const float*)d_in[8];

  char* ws = (char*)d_ws;
  const size_t MB = 1u << 20;
  unsigned short* xb   = (unsigned short*)(ws + 0 * MB);   // 8 MB [4096,1024]
  unsigned short* wqt  = (unsigned short*)(ws + 8 * MB);   // 2 MB
  unsigned short* wkt  = (unsigned short*)(ws + 10 * MB);  // 2 MB
  unsigned short* wvt  = (unsigned short*)(ws + 12 * MB);  // 2 MB
  unsigned short* wot  = (unsigned short*)(ws + 14 * MB);  // 2 MB
  unsigned short* qb   = (unsigned short*)(ws + 16 * MB);  // 8 MB [B,H,S,64]
  unsigned short* kb   = (unsigned short*)(ws + 24 * MB);  // 8 MB [B,H,S,64]
  unsigned short* vtb  = (unsigned short*)(ws + 32 * MB);  // 8 MB [B,H,64,S]
  unsigned short* valb = (unsigned short*)(ws + 40 * MB);  // 8 MB [4096,1024]

  cvt_bf16_kernel<<<(Mrows * Dd) / 256, 256, 0, stream>>>(x, xb, Mrows * Dd);
  cvt_wt_kernel<<<(Dd * Dd) / 256, 256, 0, stream>>>(Wq, wqt);
  cvt_wt_kernel<<<(Dd * Dd) / 256, 256, 0, stream>>>(Wk, wkt);
  cvt_wt_kernel<<<(Dd * Dd) / 256, 256, 0, stream>>>(Wv, wvt);
  cvt_wt_kernel<<<(Dd * Dd) / 256, 256, 0, stream>>>(Wo, wot);

  gemm_kernel<0><<<512, 256, 0, stream>>>(xb, wqt, bq, qb);
  gemm_kernel<0><<<512, 256, 0, stream>>>(xb, wkt, bk, kb);
  gemm_kernel<1><<<512, 256, 0, stream>>>(xb, wvt, bv, vtb);

  attn_kernel<<<512, 256, 0, stream>>>(qb, kb, vtb, valb);

  gemm_kernel<2><<<512, 256, 0, stream>>>(valb, wot, bo, d_out);
}
